// Attn_11373073400006
// MI455X (gfx1250) — matrix operand order repaired and checked by reading, not compiled, not run on hardware
//
#include <hip/hip_runtime.h>

// ---------------- shapes ----------------
#define BSZ   2
#define TLEN  2048
#define DIM   1024
#define NH    16
#define HD    64
#define NT    4
#define SCALE 0.125f           // 64^-0.5
#define MTOT  (BSZ * TLEN)     // 4096
#define NQKV  (3 * DIM)        // 3072

typedef __attribute__((ext_vector_type(16))) _Float16 v16h;
typedef __attribute__((ext_vector_type(8)))  _Float16 v8h;
typedef __attribute__((ext_vector_type(8)))  float    v8f;

#define WMMA_F16(a, b, c) \
  __builtin_amdgcn_wmma_f32_16x16x32_f16(false, (a), false, (b), (short)0, (c), false, false)

// A fragment (16x32 f16, row-major source, row stride ld halves).
// lane 0-15: M=lane, K {0..7,16..23}; lane 16-31: M=lane-16, K {8..15,24..31}.
__device__ __forceinline__ v16h load_a16(const _Float16* base, int ld, int m0, int k0, int lane) {
  int m = lane & 15, hi = lane >> 4;
  const _Float16* p = base + (size_t)(m0 + m) * ld + k0 + hi * 8;
  union { v16h v; v8h h[2]; } u;
  u.h[0] = *(const v8h*)(p);
  u.h[1] = *(const v8h*)(p + 16);
  return u.v;
}

// B fragment (32x16 f16) from K-contiguous source (B^T row-major, row stride ld halves).
// N = lane&15; lane 0-15: K {0..7,16..23}; lane 16-31: K {8..15,24..31}.
__device__ __forceinline__ v16h load_b16(const _Float16* base, int ld, int n0, int k0, int lane) {
  int n = lane & 15, hi = lane >> 4;
  const _Float16* p = base + (size_t)(n0 + n) * ld + k0 + hi * 8;
  union { v16h v; v8h h[2]; } u;
  u.h[0] = *(const v8h*)(p);
  u.h[1] = *(const v8h*)(p + 16);
  return u.v;
}

// reductions across the 16-lane half holding one C-fragment row
__device__ __forceinline__ float half_max16(float v) {
  #pragma unroll
  for (int m = 8; m >= 1; m >>= 1) v = fmaxf(v, __shfl_xor(v, m, 32));
  return v;
}
__device__ __forceinline__ float half_sum16(float v) {
  #pragma unroll
  for (int m = 8; m >= 1; m >>= 1) v += __shfl_xor(v, m, 32);
  return v;
}

// ---------------- kernel 1: f32 -> f16 conversion + weight fusion/transpose ----------------
__global__ __launch_bounds__(256) void convert_kernel(
    const float* __restrict__ x,
    const float* __restrict__ Wqt, const float* __restrict__ Wkt,
    const float* __restrict__ Wqs, const float* __restrict__ Wks,
    const float* __restrict__ Wv,  const float* __restrict__ Wout,
    _Float16* __restrict__ xh, _Float16* __restrict__ WqkvT, _Float16* __restrict__ WoutT) {
  size_t i = (size_t)blockIdx.x * 256 + threadIdx.x;
  const size_t NX = (size_t)MTOT * DIM;
  const size_t NW = (size_t)NQKV * DIM;
  const size_t NO = (size_t)DIM * DIM;
  if (i < NX) { xh[i] = (_Float16)x[i]; return; }
  i -= NX;
  if (i < NW) {
    int n = (int)(i >> 10), k = (int)(i & 1023);   // WqkvT[n][k]
    int sec = n >> 10, c = n & 1023, hh = c >> 6, d = c & 63;
    float v;
    if (sec == 0)       v = (hh < NT) ? Wqt[(size_t)k * (NT*HD) + hh*HD + d]
                                      : Wqs[(size_t)k * ((NH-NT)*HD) + (hh-NT)*HD + d];
    else if (sec == 1)  v = (hh < NT) ? Wkt[(size_t)k * (NT*HD) + hh*HD + d]
                                      : Wks[(size_t)k * ((NH-NT)*HD) + (hh-NT)*HD + d];
    else                v = Wv[(size_t)k * DIM + c];
    WqkvT[(size_t)n * DIM + k] = (_Float16)v;
    return;
  }
  i -= NW;
  if (i < NO) {
    int n = (int)(i >> 10), k = (int)(i & 1023);
    WoutT[(size_t)n * DIM + k] = (_Float16)Wout[(size_t)k * DIM + n];  // transpose
  }
}

// ---------------- kernel 2: fused QKV projection GEMM (WMMA, 32x64 tile, ping-pong) ----------------
__global__ __launch_bounds__(256) void gemm_qkv_kernel(
    const _Float16* __restrict__ xh, const _Float16* __restrict__ WqkvT,
    _Float16* __restrict__ Qh, _Float16* __restrict__ Kh, _Float16* __restrict__ Vt) {
  int wave = blockIdx.x * 8 + (threadIdx.x >> 5);
  int lane = threadIdx.x & 31;
  int mt = wave / (NQKV / 64), nt = wave % (NQKV / 64);
  int m0 = mt * 32, n0 = nt * 64;
  int nlo = lane & 15, hi = lane >> 4;

  v8f acc[2][4] = {{{}, {}, {}, {}}, {{}, {}, {}, {}}};
  v16h a0A, a1A, bA[4], a0B, a1B, bB[4];
  a0A = load_a16(xh, DIM, m0,      0, lane);
  a1A = load_a16(xh, DIM, m0 + 16, 0, lane);
  #pragma unroll
  for (int ns = 0; ns < 4; ns++) bA[ns] = load_b16(WqkvT, DIM, n0 + ns * 16, 0, lane);

  #pragma unroll 1
  for (int k0 = 0; k0 < DIM; k0 += 64) {        // two K-steps per iter, ping-pong frags
    __builtin_prefetch(xh + (size_t)(m0 + nlo) * DIM + k0 + 64, 0, 1);
    a0B = load_a16(xh, DIM, m0,      k0 + 32, lane);
    a1B = load_a16(xh, DIM, m0 + 16, k0 + 32, lane);
    #pragma unroll
    for (int ns = 0; ns < 4; ns++) bB[ns] = load_b16(WqkvT, DIM, n0 + ns * 16, k0 + 32, lane);
    #pragma unroll
    for (int ns = 0; ns < 4; ns++) {
      acc[0][ns] = WMMA_F16(a0A, bA[ns], acc[0][ns]);
      acc[1][ns] = WMMA_F16(a1A, bA[ns], acc[1][ns]);
    }
    if (k0 + 64 < DIM) {
      a0A = load_a16(xh, DIM, m0,      k0 + 64, lane);
      a1A = load_a16(xh, DIM, m0 + 16, k0 + 64, lane);
      #pragma unroll
      for (int ns = 0; ns < 4; ns++) bA[ns] = load_b16(WqkvT, DIM, n0 + ns * 16, k0 + 64, lane);
    }
    #pragma unroll
    for (int ns = 0; ns < 4; ns++) {
      acc[0][ns] = WMMA_F16(a0B, bB[ns], acc[0][ns]);
      acc[1][ns] = WMMA_F16(a1B, bB[ns], acc[1][ns]);
    }
  }

  #pragma unroll
  for (int mr = 0; mr < 2; mr++) {
    #pragma unroll
    for (int ns = 0; ns < 4; ns++) {
      #pragma unroll
      for (int r = 0; r < 8; r++) {
        int gm = m0 + mr * 16 + r + hi * 8;      // 0..4095 -> (b, t)
        int bb = gm >> 11, t = gm & (TLEN - 1);
        int gn = n0 + ns * 16 + nlo;             // 0..3071
        _Float16 hv = (_Float16)acc[mr][ns][r];
        if (gn < DIM) {                          // Q
          int hh = gn >> 6, d = gn & 63;
          Qh[(((size_t)(bb * NH + hh) * TLEN) + t) * HD + d] = hv;
        } else if (gn < 2 * DIM) {               // K
          int c = gn - DIM; int hh = c >> 6, d = c & 63;
          Kh[(((size_t)(bb * NH + hh) * TLEN) + t) * HD + d] = hv;
        } else {                                 // V (transposed: [b][h][d][t])
          int c = gn - 2 * DIM; int hh = c >> 6, d = c & 63;
          Vt[(((size_t)(bb * NH + hh) * HD) + d) * TLEN + t] = hv;
        }
      }
    }
  }
}

// ---------------- kernel 3: flash attention (WMMA + double-buffered async K/V in LDS) ----------------
// 8 waves/block share (b,h); each wave owns 16 query rows. K/V tiles staged via
// global_load_async_to_lds_b128 into ping-pong LDS buffers: tile i+1 loads while tile i computes.
__global__ __launch_bounds__(256) void attn_kernel(
    const _Float16* __restrict__ Qh, const _Float16* __restrict__ Kh,
    const _Float16* __restrict__ Vt, const float* __restrict__ w_sigma,
    _Float16* __restrict__ ctxh) {
  __shared__ _Float16 Ks[2][32 * 64];     // key tiles   [key][hd]   2 x 4 KB
  __shared__ _Float16 Vs[2][64 * 32];     // value tiles [hd][key]   2 x 4 KB
  __shared__ _Float16 Plds[8 * 16 * 32];  // per-wave probability tile   8 KB
  int tid  = threadIdx.x;
  int wib  = tid >> 5;
  int lane = tid & 31;
  int wave = blockIdx.x * 8 + wib;
  int qt = wave & 127;                    // T/16 = 128 query tiles
  int h  = (wave >> 7) & (NH - 1);        // uniform within a block (128 % 8 == 0)
  int bb = wave >> 11;
  int q0 = qt * 16;
  int nlo = lane & 15, hi = lane >> 4;

  float sig  = 1.f / (1.f + __expf(-w_sigma[0]));
  float coef = (h < NT) ? (-sig * SCALE) : SCALE;

  const _Float16* Qb = Qh + ((size_t)(bb * NH + h) * TLEN + q0) * HD;
  const _Float16* Kb = Kh + (size_t)(bb * NH + h) * TLEN * HD;
  const _Float16* Vb = Vt + (size_t)(bb * NH + h) * HD * TLEN;
  _Float16* Pw = Plds + wib * (16 * 32);

  v16h aQ0 = load_a16(Qb, HD, 0, 0,  lane);
  v16h aQ1 = load_a16(Qb, HD, 0, 32, lane);

  // per-thread staging slots (constant across key tiles)
  const int krow = tid >> 3, kch = tid & 7;   // K: 32 rows x 8 chunks of 8 halves
  const int vrow = tid >> 2, vch = tid & 3;   // V: 64 rows x 4 chunks of 8 halves
  unsigned kof[2], vof[2];
  kof[0] = (unsigned)(size_t)(&Ks[0][0] + krow * 64 + kch * 8);
  kof[1] = (unsigned)(size_t)(&Ks[1][0] + krow * 64 + kch * 8);
  vof[0] = (unsigned)(size_t)(&Vs[0][0] + vrow * 32 + vch * 8);
  vof[1] = (unsigned)(size_t)(&Vs[1][0] + vrow * 32 + vch * 8);

  auto issue_tile = [&](int kt, int sel) {
    unsigned long long kg =
        (unsigned long long)(size_t)(Kb + (size_t)(kt + krow) * HD + kch * 8);
    unsigned long long vg =
        (unsigned long long)(size_t)(Vb + (size_t)vrow * TLEN + kt + vch * 8);
    asm volatile("global_load_async_to_lds_b128 %0, %1, off"
                 :: "v"(kof[sel]), "v"(kg) : "memory");
    asm volatile("global_load_async_to_lds_b128 %0, %1, off"
                 :: "v"(vof[sel]), "v"(vg) : "memory");
  };

  float mrun[8], lrun[8];
  #pragma unroll
  for (int r = 0; r < 8; r++) { mrun[r] = -1e30f; lrun[r] = 0.f; }
  v8f acc[4] = {{}, {}, {}, {}};

  // prologue: stage tile 0
  issue_tile(0, 0);
  asm volatile("s_wait_asynccnt 0" ::: "memory");
  __syncthreads();

  int buf = 0;
  for (int kt = 0; kt < TLEN; kt += 32) {
    // kick off next tile into the other buffer while this one computes
    if (kt + 32 < TLEN) issue_tile(kt + 32, buf ^ 1);
    const _Float16* Kcur = &Ks[buf][0];
    const _Float16* Vcur = &Vs[buf][0];

    // ---- scores: two 16x16 tiles, contracting HD=64 in two WMMA steps (from LDS)
    v8f s0 = {}, s1 = {};
    {
      v16h b00 = load_b16(Kcur, HD, 0,  0,  lane);
      v16h b01 = load_b16(Kcur, HD, 0,  32, lane);
      s0 = WMMA_F16(aQ0, b00, s0);
      s0 = WMMA_F16(aQ1, b01, s0);
      v16h b10 = load_b16(Kcur, HD, 16, 0,  lane);
      v16h b11 = load_b16(Kcur, HD, 16, 32, lane);
      s1 = WMMA_F16(aQ0, b10, s1);
      s1 = WMMA_F16(aQ1, b11, s1);
    }
    // ---- online softmax per fragment row (row r lives across one 16-lane half)
    float alpha[8];
    #pragma unroll
    for (int r = 0; r < 8; r++) {
      float e0 = s0[r] * coef, e1 = s1[r] * coef;
      float mnew = fmaxf(mrun[r], half_max16(fmaxf(e0, e1)));
      float p0 = __expf(e0 - mnew), p1 = __expf(e1 - mnew);
      float rs = half_sum16(p0 + p1);
      alpha[r] = __expf(mrun[r] - mnew);
      lrun[r] = lrun[r] * alpha[r] + rs;
      mrun[r] = mnew;
      _Float16* pw = Pw + (r + hi * 8) * 32 + nlo;   // C layout -> [row][col] in LDS
      pw[0]  = (_Float16)p0;
      pw[16] = (_Float16)p1;
    }
    #pragma unroll
    for (int ns = 0; ns < 4; ns++)
      #pragma unroll
      for (int r = 0; r < 8; r++) acc[ns][r] *= alpha[r];

    // wave-synchronous LDS repack: wait only this wave's DS ops (no block barrier needed)
    asm volatile("s_wait_dscnt 0" ::: "memory");

    // ---- P (16x32) as A fragment from LDS
    union { v16h v; v8h h2[2]; } ua;
    const _Float16* lp = Pw + nlo * 32 + hi * 8;
    ua.h2[0] = *(const v8h*)(lp);
    ua.h2[1] = *(const v8h*)(lp + 16);

    // ---- PV: V tile in LDS is [hd][key] -> clean B fragments
    #pragma unroll
    for (int ns = 0; ns < 4; ns++) {
      v16h bv = load_b16(Vcur, 32, ns * 16, 0, lane);
      acc[ns] = WMMA_F16(ua.v, bv, acc[ns]);
    }

    // my async loads for tile i+1 done; barrier so everyone's are visible and
    // all waves have finished reading buf before it is overwritten next round
    asm volatile("s_wait_asynccnt 0" ::: "memory");
    __syncthreads();
    buf ^= 1;
  }

  #pragma unroll
  for (int r = 0; r < 8; r++) lrun[r] = 1.f / lrun[r];
  #pragma unroll
  for (int ns = 0; ns < 4; ns++) {
    #pragma unroll
    for (int r = 0; r < 8; r++) {
      int t  = q0 + r + hi * 8;
      int dg = h * HD + ns * 16 + nlo;
      ctxh[((size_t)(bb * TLEN + t)) * DIM + dg] = (_Float16)(acc[ns][r] * lrun[r]);
    }
  }
}

// ---------------- kernel 4: output projection GEMM (WMMA, 32x64 tile, ping-pong) ----------------
__global__ __launch_bounds__(256) void gemm_out_kernel(
    const _Float16* __restrict__ ctxh, const _Float16* __restrict__ WoutT,
    const float* __restrict__ bout, float* __restrict__ out) {
  int wave = blockIdx.x * 8 + (threadIdx.x >> 5);
  int lane = threadIdx.x & 31;
  int mt = wave >> 4, nt = wave & 15;           // 128 x 16 tiles of 32x64
  int m0 = mt * 32, n0 = nt * 64;
  int nlo = lane & 15, hi = lane >> 4;

  v8f acc[2][4] = {{{}, {}, {}, {}}, {{}, {}, {}, {}}};
  v16h a0A, a1A, bA[4], a0B, a1B, bB[4];
  a0A = load_a16(ctxh, DIM, m0,      0, lane);
  a1A = load_a16(ctxh, DIM, m0 + 16, 0, lane);
  #pragma unroll
  for (int ns = 0; ns < 4; ns++) bA[ns] = load_b16(WoutT, DIM, n0 + ns * 16, 0, lane);

  #pragma unroll 1
  for (int k0 = 0; k0 < DIM; k0 += 64) {
    __builtin_prefetch(ctxh + (size_t)(m0 + nlo) * DIM + k0 + 64, 0, 1);
    a0B = load_a16(ctxh, DIM, m0,      k0 + 32, lane);
    a1B = load_a16(ctxh, DIM, m0 + 16, k0 + 32, lane);
    #pragma unroll
    for (int ns = 0; ns < 4; ns++) bB[ns] = load_b16(WoutT, DIM, n0 + ns * 16, k0 + 32, lane);
    #pragma unroll
    for (int ns = 0; ns < 4; ns++) {
      acc[0][ns] = WMMA_F16(a0A, bA[ns], acc[0][ns]);
      acc[1][ns] = WMMA_F16(a1A, bA[ns], acc[1][ns]);
    }
    if (k0 + 64 < DIM) {
      a0A = load_a16(ctxh, DIM, m0,      k0 + 64, lane);
      a1A = load_a16(ctxh, DIM, m0 + 16, k0 + 64, lane);
      #pragma unroll
      for (int ns = 0; ns < 4; ns++) bA[ns] = load_b16(WoutT, DIM, n0 + ns * 16, k0 + 64, lane);
    }
    #pragma unroll
    for (int ns = 0; ns < 4; ns++) {
      acc[0][ns] = WMMA_F16(a0B, bB[ns], acc[0][ns]);
      acc[1][ns] = WMMA_F16(a1B, bB[ns], acc[1][ns]);
    }
  }

  #pragma unroll
  for (int mr = 0; mr < 2; mr++) {
    #pragma unroll
    for (int ns = 0; ns < 4; ns++) {
      #pragma unroll
      for (int r = 0; r < 8; r++) {
        int gm = m0 + mr * 16 + r + hi * 8;
        int gn = n0 + ns * 16 + nlo;
        out[(size_t)gm * DIM + gn] = acc[mr][ns][r] + bout[gn];
      }
    }
  }
}

// ---------------- host launcher ----------------
extern "C" void kernel_launch(void* const* d_in, const int* in_sizes, int n_in,
                              void* d_out, int out_size, void* d_ws, size_t ws_size,
                              hipStream_t stream) {
  const float* x    = (const float*)d_in[0];
  const float* Wqt  = (const float*)d_in[1];
  const float* Wkt  = (const float*)d_in[2];
  const float* Wqs  = (const float*)d_in[3];
  const float* Wks  = (const float*)d_in[4];
  const float* Wv   = (const float*)d_in[5];
  const float* Wout = (const float*)d_in[6];
  const float* bout = (const float*)d_in[7];
  const float* wsig = (const float*)d_in[8];
  float* out        = (float*)d_out;

  char* ws = (char*)d_ws;
  _Float16* xh    = (_Float16*)(ws);                          // 8 MB  [4096][1024]
  _Float16* WqkvT = (_Float16*)(ws + (size_t)8  * 1024*1024); // 6 MB  [3072][1024]
  _Float16* WoutT = (_Float16*)(ws + (size_t)14 * 1024*1024); // 2 MB  [1024][1024]
  _Float16* Qh    = (_Float16*)(ws + (size_t)16 * 1024*1024); // 8 MB  [B][NH][T][HD]
  _Float16* Kh    = (_Float16*)(ws + (size_t)24 * 1024*1024); // 8 MB
  _Float16* Vt    = (_Float16*)(ws + (size_t)32 * 1024*1024); // 8 MB  [B][NH][HD][T]
  _Float16* ctxh  = xh;   // xh is dead after gemm_qkv; reuse for context

  // 1) convert/pack
  {
    size_t total = (size_t)(MTOT + NQKV + DIM) * DIM;
    dim3 grid((unsigned)((total + 255) / 256));
    convert_kernel<<<grid, 256, 0, stream>>>(x, Wqt, Wkt, Wqs, Wks, Wv, Wout,
                                             xh, WqkvT, WoutT);
  }
  // 2) QKV projection: (4096/32)*(3072/64) = 6144 waves, 8 waves/block
  gemm_qkv_kernel<<<6144 / 8, 256, 0, stream>>>(xh, WqkvT, Qh, Kh, Vt);
  // 3) attention: B*NH*(T/16) = 4096 waves
  attn_kernel<<<4096 / 8, 256, 0, stream>>>(Qh, Kh, Vt, wsig, ctxh);
  // 4) output projection: (4096/32)*(1024/64) = 2048 waves
  gemm_out_kernel<<<2048 / 8, 256, 0, stream>>>(ctxh, WoutT, bout, out);
}
